// ODEModel_28441273434535
// MI455X (gfx1250) — matrix operand order repaired and checked by reading, not compiled, not run on hardware
//
#include <hip/hip_runtime.h>
#include <hip/hip_bf16.h>

// CDNA5 / gfx1250, wave32. Single-wave latency-bound RK4 scan.
// Layer1: V_WMMA_F32_16X16X4_F32 (exact f32, K=4 = the four y's; s-term + b1
//         folded into the C accumulator, shared by all 4 RK4 stages).
// Layer2: V_WMMA_F32_16X16X32_F16, b2 baked into C.
// The f32 C/D layout of layer1 matches the f16 A layout of layer2 per-lane:
// pack with v_cvt_pk_f16_f32, then ReLU in packed f16 (v_pk_max_num_f16) --
// cvt is monotonic/sign-preserving so max(0,cvt(x)) == cvt(max(0,x)).
// s_grid is staged into LDS once via the Tensor Data Mover (tensor_load_to_lds
// + s_wait_tensorcnt), with a plain staging loop after it as a correctness
// backstop (rewrites the same LDS range).

typedef float        v2f  __attribute__((ext_vector_type(2)));
typedef float        v8f  __attribute__((ext_vector_type(8)));
typedef _Float16     v16h __attribute__((ext_vector_type(16)));
typedef unsigned int u32x4 __attribute__((ext_vector_type(4)));
typedef int          i32x4 __attribute__((ext_vector_type(4)));
typedef int          i32x8 __attribute__((ext_vector_type(8)));

#if defined(__has_builtin)
#if __has_builtin(__builtin_amdgcn_wmma_f32_16x16x4_f32)
#define HAVE_WMMA_F32X4 1
#endif
#if __has_builtin(__builtin_amdgcn_tensor_load_to_lds)
#define HAVE_TDM 1
#endif
#if __has_builtin(__builtin_amdgcn_s_wait_tensorcnt)
#define HAVE_TENSORCNT 1
#endif
#if __has_builtin(__builtin_elementwise_max)
#define HAVE_ELEMMAX 1
#endif
#endif

__device__ __forceinline__ float bcast_lane(float v, int l) {
  return __int_as_float(__builtin_amdgcn_readlane(__float_as_int(v), l));
}

__device__ __forceinline__ float fast_tanh(float x) {
  // tanh(x) = 1 - 2/(e^{2x}+1); v_exp_f32 + v_rcp_f32, clamped for safety.
  float xc = __builtin_amdgcn_fmed3f(x, -15.f, 15.f);
  float e  = __expf(2.0f * xc);
  return 1.0f - 2.0f * __builtin_amdgcn_rcpf(e + 1.0f);
}

#define MAX_S 8192

__global__ void __launch_bounds__(32, 1)
ode_rk4_wmma(const float* __restrict__ S, const float* __restrict__ Y0,
             const float* __restrict__ W1, const float* __restrict__ B1,
             const float* __restrict__ W2, const float* __restrict__ B2,
             const float* __restrict__ PC, float* __restrict__ out, int n) {
  __shared__ float sgr[MAX_S];

  const int lane = threadIdx.x;        // wave32
  const int half = lane >> 4;          // 0: lanes 0-15, 1: lanes 16-31
  const int m    = lane & 15;

  const int nn = (n > MAX_S) ? MAX_S : n;

  // ---- stage s_grid into LDS via the Tensor Data Mover (one-time) ----
#if defined(HAVE_TDM) && defined(HAVE_TENSORCNT)
  {
    unsigned long long gaddr = (unsigned long long)(uintptr_t)S;
    unsigned int lds_off = (unsigned int)(uintptr_t)(&sgr[0]); // LDS byte offset

    u32x4 g0;
    g0[0] = 1u;                                      // count=1 user descriptor
    g0[1] = lds_off;                                 // lds_addr
    g0[2] = (unsigned int)(gaddr & 0xffffffffu);     // global_addr[31:0]
    g0[3] = (unsigned int)((gaddr >> 32) & 0x01ffffffu)
          | (2u << 30);                              // global_addr[56:32] | type=2

    i32x8 g1;
    g1[0] = (2 << 16);                               // data_size = 4 bytes
    g1[1] = (int)((nn & 0xffff) << 16);              // tensor_dim0[15:0]
    g1[2] = (int)(((nn >> 16) & 0xffff) | (1 << 16));// tensor_dim0[31:16], tensor_dim1=1
    g1[3] = (int)((nn & 0xffff) << 16);              // tile_dim0 = nn (<= 8192)
    g1[4] = 0;                                       // tile_dim1=0, tile_dim2=0
    g1[5] = nn;                                      // tensor_dim0_stride[31:0]
    g1[6] = 0;
    g1[7] = 0;

    i32x4 z4 = {0, 0, 0, 0};
#if __clang_major__ >= 23
    i32x8 z8 = {0, 0, 0, 0, 0, 0, 0, 0};
    __builtin_amdgcn_tensor_load_to_lds(g0, g1, z4, z4, z8, 0);
#else
    __builtin_amdgcn_tensor_load_to_lds(g0, g1, z4, z4, 0);
#endif
    __builtin_amdgcn_s_wait_tensorcnt(0);
  }
#endif
  // Correctness backstop: rewrite the same LDS range with plain loads
  // (one-time ~256 coalesced iterations; off the recurrence).
  for (int i = lane; i < nn; i += 32) sgr[i] = S[i];
  __syncthreads();

  // ---- preload constant matrices into registers ----
#if HAVE_WMMA_F32X4
  // A-layout for V_WMMA_F32_16X16X4_F32 (A = W1^T tile, 16x4):
  // lane m holds row m; VGPR0 = K (0 | 2), VGPR1 = K (1 | 3) per lane half.
  v2f A1, A2;
  A1.x = W1[(2 * half + 0) * 32 + m];
  A1.y = W1[(2 * half + 1) * 32 + m];
  A2.x = W1[(2 * half + 0) * 32 + 16 + m];
  A2.y = W1[(2 * half + 1) * 32 + 16 + m];
#else
  // VALU fallback: per-lane W1 columns producing the same D layout.
  v8f w1t1[4], w1t2[4];
#pragma unroll
  for (int k = 0; k < 4; ++k)
#pragma unroll
    for (int r = 0; r < 8; ++r) {
      w1t1[k][r] = W1[k * 32 + half * 8 + r];
      w1t2[k][r] = W1[k * 32 + 16 + half * 8 + r];
    }
#endif

  // Per-lane bias/last-row terms for the layer-1 C accumulator.
  v8f c1b, c1w, c2b, c2w;
#pragma unroll
  for (int r = 0; r < 8; ++r) {
    const int j = half * 8 + r;              // row index in D layout
    c1b[r] = B1[j];        c1w[r] = W1[4 * 32 + j];
    c2b[r] = B1[16 + j];   c2w[r] = W1[4 * 32 + 16 + j];
  }

  // B-matrix for layer 2: W2 (32x16, cols>=4 zero), f16 B layout:
  // VGPR j (j<4): K={8h+2j,8h+2j+1}; VGPR j (j>=4): K={8+8h+2j,9+8h+2j}, h=lane half.
  v16h B2h{};
  {
    const int col = m;
    const int kb  = half * 8;
#pragma unroll
    for (int j = 0; j < 8; ++j) {
      float lo = (col < 4) ? W2[(kb + 2 * j + (j >= 4 ? 8 : 0))     * 4 + col] : 0.f;
      float hi = (col < 4) ? W2[(kb + 2 * j + 1 + (j >= 4 ? 8 : 0)) * 4 + col] : 0.f;
      B2h[2 * j]     = (_Float16)lo;
      B2h[2 * j + 1] = (_Float16)hi;
    }
  }

  // Layer-2 C accumulator carries b2[n] (broadcast down each column).
  const float b2l = (m < 4) ? B2[m] : 0.f;
  v8f Cb2;
#pragma unroll
  for (int r = 0; r < 8; ++r) Cb2[r] = b2l;

  // Polynomial coefficients (uniform per lane).
  float pc[15];
#pragma unroll
  for (int i = 0; i < 15; ++i) pc[i] = PC[i];

  // Carried state (wave-uniform).
  float y0v = Y0[0], y1v = Y0[1], y2v = Y0[2], y3v = Y0[3];

  // Emit initial row.
  if (lane < 4)
    out[lane] = (lane == 0) ? y0v : (lane == 1) ? y1v : (lane == 2) ? y2v : y3v;

  // ---- one RHS evaluation: k = g(y) * tanh((relu(z@W1+b1))@W2 + b2) ----
  auto evalf = [&](float a0, float a1, float a2, float a3,
                   const v8f& C1, const v8f& C2,
                   float& k0, float& k1, float& k2, float& k3) {
#if HAVE_WMMA_F32X4
    v2f B;
    B.x = half ? a2 : a0;
    B.y = half ? a3 : a1;
    v8f D1 = __builtin_amdgcn_wmma_f32_16x16x4_f32(
        false, A1, false, B, (short)0, C1, false, false);
    v8f D2 = __builtin_amdgcn_wmma_f32_16x16x4_f32(
        false, A2, false, B, (short)0, C2, false, false);
#else
    v8f D1, D2;
#pragma unroll
    for (int r = 0; r < 8; ++r) {
      D1[r] = fmaf(a0, w1t1[0][r],
              fmaf(a1, w1t1[1][r],
              fmaf(a2, w1t1[2][r],
              fmaf(a3, w1t1[3][r], C1[r]))));
      D2[r] = fmaf(a0, w1t2[0][r],
              fmaf(a1, w1t2[1][r],
              fmaf(a2, w1t2[2][r],
              fmaf(a3, w1t2[3][r], C2[r]))));
    }
#endif
    // Pack to f16 A layout (v_cvt_pk_f16_f32), then ReLU in packed f16:
    // one v_pk_max_num_f16 per VGPR instead of 2x f32 max per element.
    v16h Ah{};
#pragma unroll
    for (int r = 0; r < 8; ++r) {
      Ah[r]     = (_Float16)D1[r];
      Ah[8 + r] = (_Float16)D2[r];
    }
#if HAVE_ELEMMAX
    {
      v16h zh{};
      Ah = __builtin_elementwise_max(Ah, zh);
    }
#else
#pragma unroll
    for (int r = 0; r < 16; ++r)
      Ah[r] = (Ah[r] > (_Float16)0) ? Ah[r] : (_Float16)0;
#endif
    v8f O = __builtin_amdgcn_wmma_f32_16x16x32_f16(
        false, Ah, false, B2h, (short)0, Cb2, false, false);

    // lane n (n<4) holds out[n]+b2[n]; activate, broadcast to all lanes.
    float t  = fast_tanh(O[0]);
    float t0 = bcast_lane(t, 0);
    float t1 = bcast_lane(t, 1);
    float t2 = bcast_lane(t, 2);
    float t3 = bcast_lane(t, 3);

    // quadratic couplings (wave-uniform; scalar-float pipe overlaps XDL path)
    float g0 = fmaf(fmaf(pc[0], a0, pc[1]), a0, pc[2]);
    float g1 = fmaf(fmaf(pc[3], a1, fmaf(pc[4], a0, pc[5])), a1, pc[6]);
    float g2 = fmaf(fmaf(pc[7], a2, fmaf(pc[8], a1, pc[9])), a2, pc[10]);
    float g3 = fmaf(fmaf(pc[11], a3, fmaf(pc[12], a2, pc[13])), a3, pc[14]);

    k0 = g0 * t0; k1 = g1 * t1; k2 = g2 * t2; k3 = g3 * t3;
  };

  // ---- sequential RK4 scan ----
  for (int i = 0; i < nn - 1; ++i) {
    const float s  = sgr[i];
    const float hh = sgr[i + 1] - s;

    // Layer-1 C accumulator: b1 + s*W1[4,:], shared by all 4 stages.
    v8f C1, C2;
#pragma unroll
    for (int r = 0; r < 8; ++r) {
      C1[r] = fmaf(s, c1w[r], c1b[r]);
      C2[r] = fmaf(s, c2w[r], c2b[r]);
    }

    float ka0, ka1, ka2, ka3, kb0, kb1, kb2, kb3;
    float kc0, kc1, kc2, kc3, kd0, kd1, kd2, kd3;

    evalf(y0v, y1v, y2v, y3v, C1, C2, ka0, ka1, ka2, ka3);
    const float hHalf = 0.5f * hh;
    evalf(fmaf(hHalf, ka0, y0v), fmaf(hHalf, ka1, y1v),
          fmaf(hHalf, ka2, y2v), fmaf(hHalf, ka3, y3v), C1, C2,
          kb0, kb1, kb2, kb3);
    evalf(fmaf(hHalf, kb0, y0v), fmaf(hHalf, kb1, y1v),
          fmaf(hHalf, kb2, y2v), fmaf(hHalf, kb3, y3v), C1, C2,
          kc0, kc1, kc2, kc3);
    evalf(fmaf(hh, kc0, y0v), fmaf(hh, kc1, y1v),
          fmaf(hh, kc2, y2v), fmaf(hh, kc3, y3v), C1, C2,
          kd0, kd1, kd2, kd3);

    const float w = hh * (1.0f / 6.0f);
    y0v = fmaf(w, ka0 + 2.f * kb0 + 2.f * kc0 + kd0, y0v);
    y1v = fmaf(w, ka1 + 2.f * kb1 + 2.f * kc1 + kd1, y1v);
    y2v = fmaf(w, ka2 + 2.f * kb2 + 2.f * kc2 + kd2, y2v);
    y3v = fmaf(w, ka3 + 2.f * kb3 + 2.f * kc3 + kd3, y3v);

    if (lane < 4) {
      float ov = (lane == 0) ? y0v : (lane == 1) ? y1v
               : (lane == 2) ? y2v : y3v;
      out[4 * (i + 1) + lane] = ov;   // off the critical path
    }
  }
}

extern "C" void kernel_launch(void* const* d_in, const int* in_sizes, int n_in,
                              void* d_out, int out_size, void* d_ws, size_t ws_size,
                              hipStream_t stream) {
  const float* S  = (const float*)d_in[0];   // s_grid (8192)
  const float* Y0 = (const float*)d_in[1];   // y0 (4)
  const float* W1 = (const float*)d_in[2];   // (5,32) row-major
  const float* B1 = (const float*)d_in[3];   // (32)
  const float* W2 = (const float*)d_in[4];   // (32,4) row-major
  const float* B2 = (const float*)d_in[5];   // (4)
  const float* PC = (const float*)d_in[6];   // (15)
  float* out = (float*)d_out;                // (8192,4)
  const int n = in_sizes[0];

  // Strictly sequential recurrence: one wave owns the whole scan.
  hipLaunchKernelGGL(ode_rk4_wmma, dim3(1), dim3(32), 0, stream,
                     S, Y0, W1, B1, W2, B2, PC, out, n);
}
